// DiffHeadGAT_5265629905484
// MI455X (gfx1250) — matrix operand order repaired and checked by reading, not compiled, not run on hardware
//
#include <hip/hip_runtime.h>
#include <stdint.h>

#define NUQ 100000
#define NIQ 50000
#define EQ  1000000

typedef __attribute__((ext_vector_type(16))) __bf16 v16bf;
typedef __attribute__((ext_vector_type(8)))  __bf16 v8bf;
typedef __attribute__((ext_vector_type(8)))  float  v8f;

#define GM_STORE 1
#define GM_BIAS  2
#define GM_DOT   4

__device__ __forceinline__ __bf16 f2bf(float f) {
    unsigned u = __float_as_uint(f);
    unsigned r = (u + 0x7FFFu + ((u >> 16) & 1u)) >> 16;
    unsigned short s = (unsigned short)r;
    __bf16 b;
    __builtin_memcpy(&b, &s, 2);
    return b;
}

// order-preserving float <-> uint for atomicMax on floats
__device__ __forceinline__ unsigned fOrd(float f) {
    unsigned u = __float_as_uint(f);
    return (u & 0x80000000u) ? ~u : (u | 0x80000000u);
}
__device__ __forceinline__ float ordF(unsigned o) {
    unsigned u = (o & 0x80000000u) ? (o & 0x7FFFFFFFu) : ~o;
    return __uint_as_float(u);
}

// ---------------- utility kernels ----------------
__global__ void fillF32K(float* p, long n, float v) {
    long i = blockIdx.x * (long)blockDim.x + threadIdx.x;
    long st = (long)blockDim.x * gridDim.x;
    for (; i < n; i += st) p[i] = v;
}
__global__ void fillU32K(unsigned* p, long n, unsigned v) {
    long i = blockIdx.x * (long)blockDim.x + threadIdx.x;
    long st = (long)blockDim.x * gridDim.x;
    for (; i < n; i += st) p[i] = v;
}
__global__ void cvtBf16K(const float* __restrict__ in, __bf16* __restrict__ out, long n) {
    long i = blockIdx.x * (long)blockDim.x + threadIdx.x;
    long st = (long)blockDim.x * gridDim.x;
    for (; i < n; i += st) out[i] = f2bf(in[i]);
}

// Pack a row-major f32 weight [K,64] into WMMA bf16 B-fragment order.
// Fragment (kt,nt): lane L, elem i  <->  B[kt*32 + (L>=16?16:0) + i][nt*16 + (L&15)]
__global__ void packWeightK(const float* __restrict__ W, __bf16* __restrict__ dst) {
    int lane = threadIdx.x;
    int kt = blockIdx.x >> 2, nt = blockIdx.x & 3;
    int kb = kt * 32 + ((lane < 16) ? 0 : 16);
    int n  = nt * 16 + (lane & 15);
#pragma unroll
    for (int i = 0; i < 16; ++i)
        dst[(size_t)(((kt * 4 + nt) * 32 + lane) * 16) + i] = f2bf(W[(size_t)(kb + i) * 64 + n]);
}

// ---------------- WMMA GEMM: C[N,64] = A[N,K](bf16) @ W(packed bf16) ----------------
// B staged into LDS per block; waves persistent over row tiles; pointer-increment
// addressing so all inner offsets are immediates. REQUIRES nRows % 16 == 0.
// NKT = K/32 (2 or 6).  LDC = compile-time output row stride.
template <int NKT, int LDC>
__global__ __launch_bounds__(256) void wmmaGemmK(
    const __bf16* __restrict__ A, int lda, int nRows,
    const __bf16* __restrict__ packB, int pBStride,
    float* __restrict__ Cout, int colOffStride,
    int mode, const float* __restrict__ bias, int biasStride,
    const float* __restrict__ dotVec, int dotVecStride,
    float* __restrict__ dotOut)
{
    __shared__ alignas(16) __bf16 bLds[NKT * 2048];   // NKT*4 fragments * 32 lanes * 16 elems

    const int lane = threadIdx.x & 31;
    const int wv = threadIdx.x >> 5;
    const int head = blockIdx.y;
    const __bf16* pB = packB + (size_t)head * pBStride;

    // cooperative stage of packed B into LDS (16B chunks)
    {
        const int chunks = NKT * 256;
        for (int c = threadIdx.x; c < chunks; c += 256)
            ((uint4*)bLds)[c] = ((const uint4*)pB)[c];
    }
    __syncthreads();

    const int m = lane & 15;
    const int hiHalf = (lane >= 16);
    const int rowAdd = hiHalf ? 8 : 0;

    // K=64: keep all 8 B fragments register-resident across the row loop
    v16bf bReg[NKT == 2 ? 8 : 1];
    if constexpr (NKT == 2) {
#pragma unroll
        for (int f = 0; f < 8; ++f)
            bReg[f] = *(const v16bf*)(bLds + (size_t)(f * 32 + lane) * 16);
    }

    // tile-invariant epilogue operands
    float biasReg[4] = {0.f, 0.f, 0.f, 0.f};
    if (mode & GM_BIAS) {
#pragma unroll
        for (int nt = 0; nt < 4; ++nt)
            biasReg[nt] = bias[head * biasStride + nt * 16 + m];
    }
    float dvReg[4] = {0.f, 0.f, 0.f, 0.f};
    if (mode & GM_DOT) {
#pragma unroll
        for (int nt = 0; nt < 4; ++nt)
            dvReg[nt] = dotVec[head * dotVecStride + nt * 16 + m];
    }

    const int tile0 = blockIdx.x * 8 + wv;
    const int tileStride = gridDim.x * 8;
    const int kOff1 = hiHalf ? 16 : 0;
    const int kOff2 = hiHalf ? 24 : 8;

    const __bf16* aPtr = A + (size_t)(tile0 * 16 + m) * lda;
    const size_t aStep = (size_t)tileStride * 16 * lda;
    float* cPtr = (mode & GM_STORE)
        ? Cout + (size_t)(tile0 * 16 + rowAdd) * LDC + head * colOffStride + m
        : nullptr;
    const size_t cStep = (size_t)tileStride * 16 * LDC;
    float* dPtr = (mode & GM_DOT)
        ? dotOut + (size_t)(tile0 * 16 + rowAdd) * 3 + head
        : nullptr;
    const size_t dStep = (size_t)tileStride * 16 * 3;

    for (int tile = tile0; tile * 16 < nRows; tile += tileStride) {
        v8f acc[4];
#pragma unroll
        for (int nt = 0; nt < 4; ++nt)
#pragma unroll
            for (int j = 0; j < 8; ++j) acc[nt][j] = 0.0f;

#pragma unroll
        for (int kt = 0; kt < NKT; ++kt) {
            v8bf c1 = *(const v8bf*)(aPtr + (kt << 5) + kOff1);
            v8bf c2 = *(const v8bf*)(aPtr + (kt << 5) + kOff2);
            v16bf af;
#pragma unroll
            for (int i = 0; i < 8; ++i) { af[i] = c1[i]; af[8 + i] = c2[i]; }
#pragma unroll
            for (int nt = 0; nt < 4; ++nt) {
                v16bf bv;
                if constexpr (NKT == 2)
                    bv = bReg[kt * 4 + nt];
                else
                    bv = *(const v16bf*)(bLds + (size_t)(((kt * 4 + nt) * 32 + lane) * 16));
                acc[nt] = __builtin_amdgcn_wmma_f32_16x16x32_bf16(
                    false, af, false, bv, (short)0, acc[nt], false, false);
            }
        }

        if (mode & GM_STORE) {
#pragma unroll
            for (int nt = 0; nt < 4; ++nt)
#pragma unroll
                for (int r = 0; r < 8; ++r)
                    cPtr[r * LDC + nt * 16] = acc[nt][r] + biasReg[nt];
            cPtr += cStep;
        }
        if (mode & GM_DOT) {
            float part[8];
#pragma unroll
            for (int r = 0; r < 8; ++r) part[r] = 0.0f;
#pragma unroll
            for (int nt = 0; nt < 4; ++nt)
#pragma unroll
                for (int r = 0; r < 8; ++r) part[r] += acc[nt][r] * dvReg[nt];
#pragma unroll
            for (int off = 1; off < 16; off <<= 1)
#pragma unroll
                for (int r = 0; r < 8; ++r) part[r] += __shfl_xor(part[r], off, 32);
            if (m == 0) {
#pragma unroll
                for (int r = 0; r < 8; ++r) dPtr[r * 3] = part[r];
            }
            dPtr += dStep;
        }
        aPtr += aStep;
    }
}

// ---------------- edge kernels ----------------
__global__ void edgeLogitMaxK(const int* __restrict__ su, const int* __restrict__ di,
    const float* __restrict__ tsU, const float* __restrict__ dgU, const float* __restrict__ dgI,
    const float* __restrict__ sSrc, const float* __restrict__ sDst,
    const float* __restrict__ at, const float* __restrict__ ap, int rel,
    float* __restrict__ elog, unsigned* __restrict__ mmax, int go, long n)
{
    float atv = at[rel], apv = ap[rel];
    long e = blockIdx.x * (long)blockDim.x + threadIdx.x;
    long st = (long)blockDim.x * gridDim.x;
    for (; e < n; e += st) {
        if (e + st < n) __builtin_prefetch(su + e + st, 0, 0);  // global_prefetch_b8
        int u = su[e], it = di[e];
        int s = go ? u : it;
        int d = go ? it : u;
        float ex1 = atv * tsU[u];
        float ex2 = apv * (dgU[u] + dgI[it]);
#pragma unroll
        for (int k = 0; k < 3; ++k) {
            float ex = (k == 0) ? 0.0f : ((k == 1) ? ex1 : ex2);
            float v = sSrc[(size_t)s * 3 + k] + sDst[(size_t)d * 3 + k] + ex;
            v = (v > 0.0f) ? v : 0.01f * v;   // leaky_relu
            elog[e * 3 + k] = v;
            atomicMax(&mmax[(size_t)d * 3 + k], fOrd(v));
        }
    }
}

__global__ void edgeExpSumK(const int* __restrict__ su, const int* __restrict__ di, int go,
    float* __restrict__ elog, const unsigned* __restrict__ mmax, float* __restrict__ den, long n)
{
    long e = blockIdx.x * (long)blockDim.x + threadIdx.x;
    long st = (long)blockDim.x * gridDim.x;
    for (; e < n; e += st) {
        int u = su[e], it = di[e];
        int d = go ? it : u;
#pragma unroll
        for (int k = 0; k < 3; ++k) {
            float mv = ordF(mmax[(size_t)d * 3 + k]);
            float ee = __expf(elog[e * 3 + k] - mv);
            elog[e * 3 + k] = ee;
            atomicAdd(&den[(size_t)d * 3 + k], ee);
        }
    }
}

__global__ void edgeAggK(const int* __restrict__ su, const int* __restrict__ di, int go,
    const float* __restrict__ elog, const float* __restrict__ den,
    const float* __restrict__ fs, float* __restrict__ head, long total)
{
    long i = blockIdx.x * (long)blockDim.x + threadIdx.x;
    long st = (long)blockDim.x * gridDim.x;
    for (; i < total; i += st) {
        long e = i / 192;
        int t = (int)(i - e * 192);
        int k = t >> 6;
        int u = su[e], it = di[e];
        int s = go ? u : it;
        int d = go ? it : u;
        float alpha = elog[e * 3 + k] / den[(size_t)d * 3 + k];
        atomicAdd(&head[(size_t)d * 192 + t], alpha * fs[(size_t)s * 192 + t]);
    }
}

// ---------------- BatchNorm(+SiLU) ----------------
__global__ void bnStatsK(const float* __restrict__ x, int n, float* __restrict__ sums) {
    int col = threadIdx.x & 63;
    int r0 = (int)((blockIdx.x * blockDim.x + threadIdx.x) >> 6);
    int rst = (int)((blockDim.x * gridDim.x) >> 6);
    float s = 0.0f, s2 = 0.0f;
    for (int r = r0; r < n; r += rst) {
        float v = x[(size_t)r * 64 + col];
        s += v; s2 += v * v;
    }
    atomicAdd(&sums[col], s);
    atomicAdd(&sums[64 + col], s2);
}
__global__ void bnApplyK(float* __restrict__ x, long n, const float* __restrict__ sums,
                         const float* __restrict__ g, const float* __restrict__ b, float invN) {
    long total = n * 64;
    long i = blockIdx.x * (long)blockDim.x + threadIdx.x;
    long st = (long)blockDim.x * gridDim.x;
    for (; i < total; i += st) {
        int col = (int)(i & 63);
        float mu = sums[col] * invN;
        float var = sums[64 + col] * invN - mu * mu;
        float rs = rsqrtf(var + 1e-5f);
        float y = (x[i] - mu) * rs * g[col] + b[col];
        x[i] = y / (1.0f + __expf(-y));   // silu
    }
}

// ---------------- host orchestration ----------------
extern "C" void kernel_launch(void* const* d_in, const int* in_sizes, int n_in,
                              void* d_out, int out_size, void* d_ws, size_t ws_size,
                              hipStream_t stream) {
    (void)in_sizes; (void)n_in; (void)out_size; (void)ws_size;
    const int NU = NUQ, NI = NIQ, E = EQ;

    const float* embU = (const float*)d_in[0];
    const float* embI = (const float*)d_in[1];
    const int*   su   = (const int*)d_in[2];
    const int*   di   = (const int*)d_in[3];
    const float* tsU  = (const float*)d_in[4];
    const float* dgU  = (const float*)d_in[5];
    const float* dgI  = (const float*)d_in[6];

    // workspace carve (256B aligned)
    char* wsb = (char*)d_ws; size_t cur = 0;
    auto take = [&](size_t bytes) -> void* {
        void* p = wsb + cur;
        cur = (cur + bytes + 255) & ~(size_t)255;
        return p;
    };
    __bf16* hbfU  = (__bf16*)take((size_t)NU * 64 * 2);
    __bf16* hbfI  = (__bf16*)take((size_t)NI * 64 * 2);
    float*  fsU   = (float*)take((size_t)NU * 192 * 4);
    float*  fsI   = (float*)take((size_t)NI * 192 * 4);
    float*  headU = (float*)take((size_t)NU * 192 * 4);
    float*  headI = (float*)take((size_t)NI * 192 * 4);
    float*  sSrcU = (float*)take((size_t)NU * 3 * 4);
    float*  sDstU = (float*)take((size_t)NU * 3 * 4);
    float*  sSrcI = (float*)take((size_t)NI * 3 * 4);
    float*  sDstI = (float*)take((size_t)NI * 3 * 4);
    unsigned* mmaxU = (unsigned*)take((size_t)NU * 3 * 4);
    float*    denU  = (float*)take((size_t)NU * 3 * 4);
    unsigned* mmaxI = (unsigned*)take((size_t)NI * 3 * 4);
    float*    denI  = (float*)take((size_t)NI * 3 * 4);
    float*  elog  = (float*)take((size_t)E * 3 * 4);
    float*  outU  = (float*)take((size_t)NU * 64 * 4);
    float*  outI  = (float*)take((size_t)NI * 64 * 4);
    float*  bnSums = (float*)take(128 * 4);
    __bf16* packbuf = (__bf16*)take(262144);   // 2 layers of packed bf16 weights

    const size_t LSTR = 61440;    // elems per layer: 12*4096 (W,Wr) + 12288 (Wc)
    // pack all weights (both layers)
    for (int l = 0; l < 2; ++l) {
        const float* W  = (const float*)d_in[7 + 11 * l];
        const float* Wr = (const float*)d_in[10 + 11 * l];
        const float* Wc = (const float*)d_in[14 + 11 * l];
        __bf16* pW  = packbuf + l * LSTR;
        __bf16* pWr = pW + 6 * 4096;
        __bf16* pWc = pW + 12 * 4096;
        for (int k = 0; k < 3; ++k)
            for (int r = 0; r < 2; ++r) {
                int idx = k * 2 + r;
                packWeightK<<<8, 32, 0, stream>>>(W + (size_t)idx * 4096, pW + (size_t)idx * 4096);
                packWeightK<<<8, 32, 0, stream>>>(Wr + (size_t)idx * 4096, pWr + (size_t)idx * 4096);
            }
        packWeightK<<<24, 32, 0, stream>>>(Wc, pWc);
    }

    // persistent-wave GEMM: cap blocks so each wave loops over >1 tile
    auto capBlocks = [](int n) {
        int tiles8 = (n + 127) / 128;
        return tiles8 < 256 ? tiles8 : 256;
    };
    auto gemm64 = [&](const __bf16* A, int lda, int n, const __bf16* pB, int pBs,
                      float* C, int colStr, int mode,
                      const float* bias, int biasStr,
                      const float* dv, int dvStr, float* dOut, int heads) {
        dim3 g(capBlocks(n), heads);
        wmmaGemmK<2, 192><<<g, 256, 0, stream>>>(A, lda, n, pB, pBs, C, colStr,
                                                 mode, bias, biasStr, dv, dvStr, dOut);
    };
    auto gemm192 = [&](const __bf16* A, int n, const __bf16* pB,
                       float* C, int mode, const float* bias) {
        dim3 g(capBlocks(n), 1);
        wmmaGemmK<6, 64><<<g, 256, 0, stream>>>(A, 192, n, pB, 0, C, 0,
                                                mode, bias, 0, nullptr, 0, nullptr);
    };

    float* doutU = (float*)d_out;
    float* doutI = doutU + (size_t)NU * 64;

    auto layer = [&](int l, const float* hu, const float* hi, float* oU, float* oI) {
        int pb = 7 + 11 * l;
        const float* al  = (const float*)d_in[pb + 1];
        const float* ar  = (const float*)d_in[pb + 2];
        const float* bb  = (const float*)d_in[pb + 4];
        const float* at  = (const float*)d_in[pb + 5];
        const float* ap  = (const float*)d_in[pb + 6];
        const float* bc  = (const float*)d_in[pb + 8];
        const float* bng = (const float*)d_in[pb + 9];
        const float* bnb = (const float*)d_in[pb + 10];
        __bf16* pW  = packbuf + l * LSTR;
        __bf16* pWr = pW + 6 * 4096;
        __bf16* pWc = pW + 12 * 4096;

        cvtBf16K<<<4096, 256, 0, stream>>>(hu, hbfU, (long)NU * 64);
        cvtBf16K<<<4096, 256, 0, stream>>>(hi, hbfI, (long)NI * 64);

        // projections: fs (stored) + logit-scalar dots (al); fd only as dot (ar)
        gemm64(hbfU, 64, NU, pW,        8192, fsU, 64, GM_STORE | GM_DOT,
               nullptr, 0, al,      128, sSrcU, 3);
        gemm64(hbfI, 64, NI, pW,        8192, nullptr, 0, GM_DOT,
               nullptr, 0, ar,      128, sDstI, 3);
        gemm64(hbfI, 64, NI, pW + 4096, 8192, fsI, 64, GM_STORE | GM_DOT,
               nullptr, 0, al + 64, 128, sSrcI, 3);
        gemm64(hbfU, 64, NU, pW + 4096, 8192, nullptr, 0, GM_DOT,
               nullptr, 0, ar + 64, 128, sDstU, 3);
        // residual: head = h_dst @ Wr + b  (overwrite, then edges accumulate on top)
        gemm64(hbfI, 64, NI, pWr,        8192, headI, 64, GM_STORE | GM_BIAS,
               bb,      128, nullptr, 0, nullptr, 3);
        gemm64(hbfU, 64, NU, pWr + 4096, 8192, headU, 64, GM_STORE | GM_BIAS,
               bb + 64, 128, nullptr, 0, nullptr, 3);

        fillU32K<<<600, 256, 0, stream>>>(mmaxI, (long)NI * 3, 0u);
        fillF32K<<<600, 256, 0, stream>>>(denI,  (long)NI * 3, 0.0f);
        fillU32K<<<1200, 256, 0, stream>>>(mmaxU, (long)NU * 3, 0u);
        fillF32K<<<1200, 256, 0, stream>>>(denU,  (long)NU * 3, 0.0f);

        const long aggTot = (long)E * 192;
        const int eB = (E + 255) / 256;
        const unsigned aB = (unsigned)((aggTot + 255) / 256);
        // relation 'go': user -> item (dst = items)
        edgeLogitMaxK<<<eB, 256, 0, stream>>>(su, di, tsU, dgU, dgI, sSrcU, sDstI,
                                              at, ap, 0, elog, mmaxI, 1, (long)E);
        edgeExpSumK<<<eB, 256, 0, stream>>>(su, di, 1, elog, mmaxI, denI, (long)E);
        edgeAggK<<<aB, 256, 0, stream>>>(su, di, 1, elog, denI, fsU, headI, aggTot);
        // relation 'back': item -> user (dst = users)
        edgeLogitMaxK<<<eB, 256, 0, stream>>>(su, di, tsU, dgU, dgI, sSrcI, sDstU,
                                              at, ap, 1, elog, mmaxU, 0, (long)E);
        edgeExpSumK<<<eB, 256, 0, stream>>>(su, di, 0, elog, mmaxU, denU, (long)E);
        edgeAggK<<<aB, 256, 0, stream>>>(su, di, 0, elog, denU, fsI, headU, aggTot);

        // head concat @ Wc + bc  (reuse fs buffers as bf16 staging: fs no longer needed)
        __bf16* hbU2 = (__bf16*)fsU;
        __bf16* hbI2 = (__bf16*)fsI;
        cvtBf16K<<<8192, 256, 0, stream>>>(headU, hbU2, (long)NU * 192);
        cvtBf16K<<<8192, 256, 0, stream>>>(headI, hbI2, (long)NI * 192);
        gemm192(hbU2, NU, pWc, oU, GM_STORE | GM_BIAS, bc);
        gemm192(hbI2, NI, pWc, oI, GM_STORE | GM_BIAS, bc);

        // BN + SiLU (users: row 0 of bng/bnb; items: row 1)
        fillF32K<<<1, 128, 0, stream>>>(bnSums, 128, 0.0f);
        bnStatsK<<<256, 256, 0, stream>>>(oU, NU, bnSums);
        bnApplyK<<<8192, 256, 0, stream>>>(oU, (long)NU, bnSums, bng, bnb, 1.0f / NU);
        fillF32K<<<1, 128, 0, stream>>>(bnSums, 128, 0.0f);
        bnStatsK<<<256, 256, 0, stream>>>(oI, NI, bnSums);
        bnApplyK<<<8192, 256, 0, stream>>>(oI, (long)NI, bnSums, bng + 64, bnb + 64, 1.0f / NI);
    };

    layer(0, embU, embI, outU, outI);
    layer(1, outU, outI, doutU, doutI);
}
